// DynaNetGNN_83099027243014
// MI455X (gfx1250) — matrix operand order repaired and checked by reading, not compiled, not run on hardware
//
#include <hip/hip_runtime.h>

typedef _Float16 f16;
typedef __attribute__((ext_vector_type(16))) _Float16 v16h;
typedef __attribute__((ext_vector_type(8)))  float    v8f;

#define TSEQ 50

static inline unsigned gblk(long n, int b = 256) { return (unsigned)((n + b - 1) / b); }

// ---------------- elementwise helpers ----------------
__global__ __launch_bounds__(256) void k_f32_to_f16(const float* __restrict__ x, f16* __restrict__ y, long n) {
  long i = (long)blockIdx.x * blockDim.x + threadIdx.x;
  if (i < n) y[i] = (f16)x[i];
}

__global__ __launch_bounds__(256) void k_relu_f16(const float* __restrict__ x, f16* __restrict__ y, long n) {
  long i = (long)blockIdx.x * blockDim.x + threadIdx.x;
  if (i < n) y[i] = (f16)fmaxf(x[i], 0.f);
}

// transpose weights K x N (f32) -> N x ldw (f16), zero-padded to ldw
__global__ __launch_bounds__(256) void k_w16t(const float* __restrict__ w, f16* __restrict__ out,
                                              int K, int N, int ldw) {
  long total = (long)N * ldw;
  long id = (long)blockIdx.x * blockDim.x + threadIdx.x;
  if (id >= total) return;
  int k = (int)(id % ldw);
  long n = id / ldw;
  out[id] = (k < K) ? (f16)w[(long)k * N + n] : (f16)0.f;
}

// ---------------- WMMA GEMM: C[M x N] = act(A[M x K] * W[K x N] + bias) ----------------
// A: f16 row-major, leading dim lda (multiple of 8 whenever K >= 32).
// Wt: f16 TRANSPOSED weights, N rows x ldw cols (ldw = K rounded up to 16), zero padded.
// One wave computes a 16x32 output tile (two 16x16 WMMA tiles sharing the A fragment).
// Requires M % 16 == 0.
union U16x8 { uint4 q; f16 h[8]; };

__global__ __launch_bounds__(32) void k_gemm_f16(const f16* __restrict__ A, long lda,
                                                 const f16* __restrict__ Wt, long ldw,
                                                 const float* __restrict__ bias,
                                                 float* __restrict__ C, long ldc,
                                                 int M, int N, int K, int relu) {
  const int lane = threadIdx.x;
  const int half = lane >> 4;
  const int l16  = lane & 15;
  const long m0 = (long)blockIdx.y * 16;
  const long n0 = (long)blockIdx.x * 32;
  const long ncol0 = n0 + l16;
  const long ncol1 = n0 + 16 + l16;
  const long nc0 = (ncol0 < N) ? ncol0 : (long)(N - 1); // clamp: garbage cols never stored
  const long nc1 = (ncol1 < N) ? ncol1 : (long)(N - 1);
  const long arow = (m0 + l16) * lda;
  v8f acc0 = {}, acc1 = {};

  const int kmain = K & ~31;
  for (int k0 = 0; k0 < kmain; k0 += 32) {
    // A fragment: lanes 0-15 hold K k0+0..15, lanes 16-31 hold k0+16..31
    const f16* ap = A + arow + k0 + half * 16;
    U16x8 a0, a1;
    a0.q = *(const uint4*)(ap);
    a1.q = *(const uint4*)(ap + 8);
    // B fragments: lanes 0-15 hold K k0+0..15, lanes 16-31 hold k0+16..31 (contiguous in Wt)
    const f16* bp0 = Wt + nc0 * ldw + k0 + half * 16;
    const f16* bp1 = Wt + nc1 * ldw + k0 + half * 16;
    U16x8 b0a, b0b, b1a, b1b;
    b0a.q = *(const uint4*)(bp0); b0b.q = *(const uint4*)(bp0 + 8);
    b1a.q = *(const uint4*)(bp1); b1b.q = *(const uint4*)(bp1 + 8);
    v16h a, b0, b1;
#pragma unroll
    for (int e = 0; e < 8; ++e) {
      a[e] = a0.h[e]; a[8 + e] = a1.h[e];
      b0[e] = b0a.h[e]; b0[8 + e] = b0b.h[e];
      b1[e] = b1a.h[e]; b1[8 + e] = b1b.h[e];
    }
    acc0 = __builtin_amdgcn_wmma_f32_16x16x32_f16(false, a, false, b0, (short)0, acc0, false, false);
    acc1 = __builtin_amdgcn_wmma_f32_16x16x32_f16(false, a, false, b1, (short)0, acc1, false, false);
  }

  if (kmain < K) { // tail: clamp index, select zero (branch-free)
    v16h a, b0, b1;
#pragma unroll
    for (int e = 0; e < 16; ++e) {
      int ka = kmain + half * 16 + e;
      f16 av = A[arow + (ka < K ? ka : 0)];
      a[e] = (ka < K) ? av : (f16)0.f;
      int kb = kmain + half * 16 + e;
      int kbc = (kb < K) ? kb : 0;
      f16 bv0 = Wt[nc0 * ldw + kbc];
      f16 bv1 = Wt[nc1 * ldw + kbc];
      b0[e] = (kb < K) ? bv0 : (f16)0.f;
      b1[e] = (kb < K) ? bv1 : (f16)0.f;
    }
    acc0 = __builtin_amdgcn_wmma_f32_16x16x32_f16(false, a, false, b0, (short)0, acc0, false, false);
    acc1 = __builtin_amdgcn_wmma_f32_16x16x32_f16(false, a, false, b1, (short)0, acc1, false, false);
  }

  if (ncol0 < N) {
    float bv = bias ? bias[ncol0] : 0.f;
#pragma unroll
    for (int r = 0; r < 8; ++r) {
      long m = m0 + r + 8 * half;
      float v = acc0[r] + bv;
      if (relu) v = fmaxf(v, 0.f);
      C[m * ldc + ncol0] = v;
    }
  }
  if (ncol1 < N) {
    float bv = bias ? bias[ncol1] : 0.f;
#pragma unroll
    for (int r = 0; r < 8; ++r) {
      long m = m0 + r + 8 * half;
      float v = acc1[r] + bv;
      if (relu) v = fmaxf(v, 0.f);
      C[m * ldc + ncol1] = v;
    }
  }
}

// ---------------- train-mode BatchNorm1d (stats over rows) ----------------
__global__ __launch_bounds__(256) void k_colstats(const float* __restrict__ X, int M, int Ncols,
                                                  float* __restrict__ mean, float* __restrict__ rstd) {
  int j = blockIdx.x;
  __shared__ float s1[256], s2[256];
  float a = 0.f, b = 0.f;
  for (int i = threadIdx.x; i < M; i += 256) {
    float v = X[(long)i * Ncols + j];
    a += v; b += v * v;
  }
  s1[threadIdx.x] = a; s2[threadIdx.x] = b;
  __syncthreads();
  for (int s = 128; s > 0; s >>= 1) {
    if (threadIdx.x < (unsigned)s) { s1[threadIdx.x] += s1[threadIdx.x + s]; s2[threadIdx.x] += s2[threadIdx.x + s]; }
    __syncthreads();
  }
  if (threadIdx.x == 0) {
    float m = s1[0] / (float)M;
    float v = s2[0] / (float)M - m * m;
    mean[j] = m;
    rstd[j] = rsqrtf(v + 1e-5f);
  }
}

__global__ __launch_bounds__(256) void k_bn_act(const float* __restrict__ X,
                                                const float* __restrict__ mean, const float* __restrict__ rstd,
                                                const float* __restrict__ g, const float* __restrict__ be,
                                                long total, int Ncols, int relu,
                                                float* __restrict__ out32, f16* __restrict__ out16) {
  long i = (long)blockIdx.x * blockDim.x + threadIdx.x;
  if (i >= total) return;
  int j = (int)(i % Ncols);
  float v = (X[i] - mean[j]) * rstd[j] * g[j] + be[j];
  if (relu) v = fmaxf(v, 0.f);
  if (out32) out32[i] = v;
  if (out16) out16[i] = (f16)v;
}

// ---------------- graph concat / gather builders ----------------
__global__ __launch_bounds__(256) void k_build_edge4(const float* __restrict__ kp, f16* __restrict__ out) {
  long id = (long)blockIdx.x * blockDim.x + threadIdx.x;
  if (id >= 400L * 256) return;
  int j = (int)(id & 15), i = (int)((id >> 4) & 15);
  long g = id >> 8;
  const float* pi = kp + ((g * 16 + i) * 2);
  const float* pj = kp + ((g * 16 + j) * 2);
  f16* o = out + id * 4;
  o[0] = (f16)pi[0]; o[1] = (f16)pi[1]; o[2] = (f16)pj[0]; o[3] = (f16)pj[1];
}

// [nf_i(F), nf_j(F), ef_ij(F)] -> (G*256, 3F)
__global__ __launch_bounds__(256) void k_concat3(const f16* __restrict__ nf, const f16* __restrict__ ef,
                                                 f16* __restrict__ out, int G, int F) {
  long total = (long)G * 256 * 3 * F;
  long id = (long)blockIdx.x * blockDim.x + threadIdx.x;
  if (id >= total) return;
  int C3 = 3 * F;
  int c = (int)(id % C3);
  long row = id / C3;
  int j = (int)(row & 15), i = (int)((row >> 4) & 15);
  long g = row >> 8;
  f16 v;
  if (c < F)          v = nf[(g * 16 + i) * F + c];
  else if (c < 2 * F) v = nf[(g * 16 + j) * F + (c - F)];
  else                v = ef[row * F + (c - 2 * F)];
  out[id] = v;
}

// node propagator input: [ne(96), node_eff(96), sum_j edge_eff(96)] -> (G*16, 288)
__global__ __launch_bounds__(256) void k_concat_np(const f16* __restrict__ ne, const f16* __restrict__ neff,
                                                   const f16* __restrict__ ef, f16* __restrict__ out, int G) {
  long total = (long)G * 16 * 288;
  long id = (long)blockIdx.x * blockDim.x + threadIdx.x;
  if (id >= total) return;
  int c = (int)(id % 288);
  long node = id / 288;
  f16 v;
  if (c < 96)       v = ne[node * 96 + c];
  else if (c < 192) v = neff[node * 96 + (c - 96)];
  else {
    int cc = c - 192;
    const f16* p = ef + node * 16 * 96 + cc;
    float s = 0.f;
#pragma unroll
    for (int j = 0; j < 16; ++j) s += (float)p[(long)j * 96];
    v = (f16)s;
  }
  out[id] = v;
}

// [a(F), b(F)] -> (rows, 2F)
__global__ __launch_bounds__(256) void k_concat2(const f16* __restrict__ a, const f16* __restrict__ b,
                                                 f16* __restrict__ out, long rows, int F) {
  long total = rows * 2 * F;
  long id = (long)blockIdx.x * blockDim.x + threadIdx.x;
  if (id >= total) return;
  int c = (int)(id % (2 * F));
  long r = id / (2 * F);
  out[id] = (c < F) ? a[r * F + c] : b[r * F + (c - F)];
}

// GRU sequence inputs; node input padded 36 -> stride 40 (16B row alignment for b128 loads)
__global__ __launch_bounds__(256) void k_build_node_in(const float* __restrict__ nrep, const float* __restrict__ kp,
                                                       const float* __restrict__ act, f16* __restrict__ out) {
  long total = 128L * TSEQ * 40;
  long id = (long)blockIdx.x * blockDim.x + threadIdx.x;
  if (id >= total) return;
  int c = (int)(id % 40);
  long r = id / 40;
  int t = (int)(r % TSEQ); r /= TSEQ;
  int i = (int)(r % 16);
  int b = (int)(r / 16);
  long base = ((long)b * TSEQ + t) * 16 + i;
  float v;
  if (c < 32)      v = nrep[base * 32 + c];
  else if (c < 34) v = kp[base * 2 + (c - 32)];
  else if (c < 36) v = act[base * 2 + (c - 34)];
  else             v = 0.f;
  out[id] = (f16)v;
}

__global__ __launch_bounds__(256) void k_build_edge_in(const float* __restrict__ erep, const float* __restrict__ kp,
                                                       const float* __restrict__ act, f16* __restrict__ out) {
  long total = 2048L * TSEQ * 40;
  long id = (long)blockIdx.x * blockDim.x + threadIdx.x;
  if (id >= total) return;
  int c = (int)(id % 40);
  long r = id / 40;
  int t = (int)(r % TSEQ); r /= TSEQ;
  int j = (int)(r % 16); r /= 16;
  int i = (int)(r % 16);
  int b = (int)(r / 16);
  long bt = (long)b * TSEQ + t;
  long bi = bt * 16 + i;
  long bj = bt * 16 + j;
  float v;
  if (c < 32)      v = erep[((bt * 16 + i) * 16 + j) * 32 + c];
  else if (c < 34) v = kp[bi * 2 + (c - 32)];
  else if (c < 36) v = kp[bj * 2 + (c - 34)];
  else if (c < 38) v = act[bi * 2 + (c - 36)];
  else             v = act[bj * 2 + (c - 38)];
  out[id] = (f16)v;
}

// ---------------- GRU gate math ----------------
__global__ __launch_bounds__(256) void k_gru_gates(const float* __restrict__ gi, const float* __restrict__ gh,
                                                   float* __restrict__ h, f16* __restrict__ h16,
                                                   f16* __restrict__ hseq, int rows, int t, int T) {
  long total = (long)rows * 128;
  long id = (long)blockIdx.x * blockDim.x + threadIdx.x;
  if (id >= total) return;
  int c = (int)(id % 128);
  long r = id / 128;
  long b = r * 384;
  float ir = gi[b + c],       hr = gh[b + c];
  float iz = gi[b + 128 + c], hz = gh[b + 128 + c];
  float in = gi[b + 256 + c], hn = gh[b + 256 + c];
  float rg = 1.f / (1.f + expf(-(ir + hr)));
  float z  = 1.f / (1.f + expf(-(iz + hz)));
  float n  = tanhf(in + rg * hn);
  float hv = (1.f - z) * n + z * h[id];
  h[id] = hv;
  h16[id] = (f16)hv;
  hseq[(r * T + t) * 128 + c] = (f16)hv;
}

// ---------------- gumbel softmax on logits ----------------
__device__ inline float hash_u01(unsigned x) {
  x ^= x >> 17; x *= 0xed5ad4bbu; x ^= x >> 11; x *= 0xac4c1b51u;
  x ^= x >> 15; x *= 0x31848babu; x ^= x >> 14;
  return ((float)(x >> 8) + 0.5f) * (1.0f / 16777216.0f);
}

__global__ __launch_bounds__(256) void k_gumbel(const float* __restrict__ lraw,
                                                float* __restrict__ out_lg, float* __restrict__ out_et) {
  int id = blockIdx.x * blockDim.x + threadIdx.x;
  if (id >= 8 * 16 * 16) return;
  int j = id & 15, i = (id >> 4) & 15, b = id >> 8;
  int fwd = id * 3;
  int rev = ((b * 16 + j) * 16 + i) * 3;
  float p[3], mx = -1e30f;
  for (int k = 0; k < 3; ++k) {
    float l = 0.5f * (lraw[fwd + k] + lraw[rev + k]);
    out_lg[fwd + k] = l;
    float u = hash_u01((unsigned)(fwd + k) * 2654435761u + 42u);
    u = fminf(fmaxf(u, 1e-6f), 1.f - 1e-6f);
    float gn = -logf(-logf(u + 1e-10f) + 1e-10f);
    p[k] = (l + gn) * 2.0f; // / TEMP (0.5)
    mx = fmaxf(mx, p[k]);
  }
  float s = 0.f;
  for (int k = 0; k < 3; ++k) { p[k] = expf(p[k] - mx); s += p[k]; }
  float mask = (i == j) ? 0.f : 1.f;
  for (int k = 0; k < 3; ++k) out_et[fwd + k] = mask * p[k] / s;
}

// edge-type weighted sum of 3 encoder/propagator outputs (2048 x 96)
__global__ __launch_bounds__(256) void k_wsum3(const f16* __restrict__ e0, const f16* __restrict__ e1,
                                               const f16* __restrict__ e2, const float* __restrict__ et,
                                               f16* __restrict__ out) {
  long total = 2048L * 96;
  long id = (long)blockIdx.x * blockDim.x + threadIdx.x;
  if (id >= total) return;
  long row = id / 96;
  float v = et[row * 3 + 0] * (float)e0[id] + et[row * 3 + 1] * (float)e1[id] + et[row * 3 + 2] * (float)e2[id];
  out[id] = (f16)v;
}

__global__ __launch_bounds__(256) void k_sym4(const float* __restrict__ raw, float* __restrict__ out) {
  long id = (long)blockIdx.x * blockDim.x + threadIdx.x;
  if (id >= 8L * 256 * 4) return;
  int c = (int)(id & 3);
  long r = id >> 2;
  int j = (int)(r & 15), i = (int)((r >> 4) & 15);
  long b = r >> 8;
  long rev = (((b * 16 + j) * 16 + i) << 2) + c;
  out[id] = 0.5f * (raw[id] + raw[rev]);
}

// =====================================================================
extern "C" void kernel_launch(void* const* d_in, const int* in_sizes, int n_in,
                              void* d_out_v, int out_size, void* d_ws, size_t ws_size,
                              hipStream_t stream) {
  (void)in_sizes; (void)n_in; (void)out_size; (void)ws_size;
  float* d_out = (float*)d_out_v;
  float* out_na = d_out;            // (8,16,4)
  float* out_ea = d_out + 512;      // (8,16,16,4)
  float* out_et = d_out + 8704;     // (8,16,16,3)
  float* out_lg = d_out + 14848;    // (8,16,16,3)

  int idx = 0;
  const float* kp     = (const float*)d_in[idx++];
  const float* action = (const float*)d_in[idx++];

  struct LinP { const float* w; const float* b; };
  struct BnP  { const float* g; const float* be; };
  struct PropP {
    LinP ne; BnP ne_bn;
    LinP ee_l[3]; BnP ee_bn[3];
    LinP np_l1, np_l2; BnP np_bn;
    LinP ep_l1[2][3], ep_l2[2][3]; BnP ep_bn[2][3];
    LinP npred_l1, npred_l2; BnP npred_bn;
    LinP epred_l1, epred_l2; BnP epred_bn;
  };
  struct GruP { const float* wih[2]; const float* whh[2]; const float* bih[2]; const float* bhh[2]; LinP fc; };

  auto nextp = [&]() -> const float* { return (const float*)d_in[idx++]; };
  auto lin = [&]() -> LinP { LinP p; p.w = nextp(); p.b = nextp(); return p; };
  auto bnp = [&]() -> BnP { BnP p; p.g = nextp(); p.be = nextp(); return p; };
  auto prop = [&](int etn, int pstep, bool has_np) -> PropP {
    PropP p{};
    p.ne = lin(); p.ne_bn = bnp();
    for (int k = 0; k < etn; ++k) { p.ee_l[k] = lin(); p.ee_bn[k] = bnp(); }
    p.np_l1 = lin(); p.np_l2 = lin(); p.np_bn = bnp();
    for (int i = 0; i < pstep; ++i)
      for (int k = 0; k < etn; ++k) { p.ep_l1[i][k] = lin(); p.ep_l2[i][k] = lin(); p.ep_bn[i][k] = bnp(); }
    if (has_np) { p.npred_l1 = lin(); p.npred_bn = bnp(); p.npred_l2 = lin(); }
    p.epred_l1 = lin(); p.epred_bn = bnp(); p.epred_l2 = lin();
    return p;
  };
  auto grup = [&]() -> GruP {
    GruP p{};
    for (int l = 0; l < 2; ++l) { p.wih[l] = nextp(); p.whh[l] = nextp(); p.bih[l] = nextp(); p.bhh[l] = nextp(); }
    p.fc = lin();
    return p;
  };

  PropP enc  = prop(1, 1, true);
  GruP  ngru = grup();
  GruP  egru = grup();
  PropP affi = prop(1, 2, false);
  PropP attr = prop(3, 1, true);

  // ---------- workspace arena ----------
  size_t off = 0;
  auto alloc = [&](size_t bytes) -> void* {
    size_t a = (off + 255) & ~(size_t)255;
    off = a + bytes;
    return (void*)((char*)d_ws + a);
  };

  const long RE = 102400; // B*T*N*N
  const long RN = 6400;   // B*T*N

  float* stat_m = (float*)alloc(96 * 4);
  float* stat_r = (float*)alloc(96 * 4);
  float* BB = (float*)alloc(RE * 96 * 4);   // GEMM out pool 1
  float* BC = (float*)alloc(RE * 96 * 4);   // GEMM out pool 2
  f16*   BD = (f16*)alloc(RE * 96 * 2);     // f16 scratch pool
  f16*   BA = (f16*)alloc(RE * 288 * 2);    // concat input pool

  f16* ne16    = (f16*)alloc(RN * 96 * 2);
  f16* neff16  = (f16*)alloc(RN * 96 * 2);
  f16* eenc16  = (f16*)alloc(RE * 96 * 2);
  f16* eeff16  = (f16*)alloc(RE * 96 * 2);
  float* nrep  = (float*)alloc(RN * 32 * 4);
  float* erep  = (float*)alloc(RE * 32 * 4);
  f16* nin16   = (f16*)alloc(128L * TSEQ * 40 * 2);
  f16* ein16   = (f16*)alloc(2048L * TSEQ * 40 * 2);
  f16* nhseq   = (f16*)alloc(128L * TSEQ * 128 * 2);
  f16* ehseq   = (f16*)alloc(2048L * TSEQ * 128 * 2);
  float* hbuf  = (float*)alloc(2048L * 128 * 4);
  f16*   h16   = (f16*)alloc(2048L * 128 * 2);
  float* gi    = (float*)alloc(2048L * 384 * 4);
  float* gh    = (float*)alloc(2048L * 384 * 4);
  float* nagg  = (float*)alloc(128L * 32 * 4);
  float* eagg  = (float*)alloc(2048L * 32 * 4);
  f16* nagg16  = (f16*)alloc(128L * 32 * 2);
  f16* eagg16  = (f16*)alloc(2048L * 32 * 2);
  f16* eein16  = (f16*)alloc(2048L * 96 * 2);
  f16* a_ne16  = (f16*)alloc(128L * 96 * 2);
  f16* a_neff  = (f16*)alloc(128L * 96 * 2);
  f16* a_eenc  = (f16*)alloc(2048L * 96 * 2);
  f16* a_eeff  = (f16*)alloc(2048L * 96 * 2);
  float* lraw  = (float*)alloc(2048L * 3 * 4);
  f16* ek0     = (f16*)alloc(2048L * 96 * 2);
  f16* ek1     = (f16*)alloc(2048L * 96 * 2);
  f16* ek2     = (f16*)alloc(2048L * 96 * 2);
  f16* t_ne16  = (f16*)alloc(128L * 96 * 2);
  f16* t_neff  = (f16*)alloc(128L * 96 * 2);
  f16* t_eenc  = (f16*)alloc(2048L * 96 * 2);
  f16* t_eeff  = (f16*)alloc(2048L * 96 * 2);
  float* eraw  = (float*)alloc(2048L * 4 * 4);

  // ---------- host-side helpers ----------
  // convert + transpose weights: (K x N f32) -> (N x ldw f16), ldw = roundup(K,16)
  auto W16 = [&](const float* w, int K, int N) -> f16* {
    int ldw = (K + 15) & ~15;
    f16* p = (f16*)alloc((size_t)N * ldw * 2);
    k_w16t<<<gblk((long)N * ldw), 256, 0, stream>>>(w, p, K, N, ldw);
    return p;
  };
  auto gemm = [&](const f16* A, long lda, const f16* Wt, int Nc, const float* bias,
                  float* C, long ldc, long M, int K, int relu) {
    long ldw = (K + 15) & ~15;
    dim3 grid((unsigned)((Nc + 31) / 32), (unsigned)((M + 15) / 16));
    k_gemm_f16<<<grid, 32, 0, stream>>>(A, lda, Wt, ldw, bias, C, ldc, (int)M, Nc, K, relu);
  };
  auto colbn = [&](const float* X, long M, BnP bn, int relu, f16* o16, float* o32) {
    k_colstats<<<96, 256, 0, stream>>>(X, (int)M, 96, stat_m, stat_r);
    long total = M * 96;
    k_bn_act<<<gblk(total), 256, 0, stream>>>(X, stat_m, stat_r, bn.g, bn.be, total, 96, relu, o32, o16);
  };
  // Linear-ReLU-Linear-ReLU-BN (PropNet propagator)
  auto mlp2 = [&](const f16* x16, long M, int Kin, LinP l1, LinP l2, BnP bn, f16* out16) {
    f16* w1 = W16(l1.w, Kin, 96);
    gemm(x16, Kin, w1, 96, l1.b, BB, 96, M, Kin, 1);
    k_f32_to_f16<<<gblk(M * 96), 256, 0, stream>>>(BB, BD, M * 96);
    f16* w2 = W16(l2.w, 96, 96);
    gemm(BD, 96, w2, 96, l2.b, BC, 96, M, 96, 1);
    colbn(BC, M, bn, 0, out16, nullptr);
  };
  // Linear-BN-ReLU-Linear (PropNet predictor)
  auto predf = [&](const f16* x16, long M, LinP l1, BnP bn, LinP l2, int dout, float* out, long ldc) {
    f16* w1 = W16(l1.w, 192, 96);
    gemm(x16, 192, w1, 96, l1.b, BB, 96, M, 192, 0);
    colbn(BB, M, bn, 1, BD, nullptr);
    f16* w2 = W16(l2.w, 96, dout);
    gemm(BD, 96, w2, dout, l2.b, out, ldc, M, 96, 0);
  };
  auto rungru = [&](const f16* X16, int rows, int din, int dstride, GruP& g, f16* hseq, float* o32) {
    const f16* xin = X16;
    int xk = din, xs = dstride;
    long xlda = (long)TSEQ * dstride;
    for (int layer = 0; layer < 2; ++layer) {
      f16* wih = W16(g.wih[layer], xk, 384);
      f16* whh = W16(g.whh[layer], 128, 384);
      hipMemsetAsync(hbuf, 0, (size_t)rows * 128 * 4, stream);
      hipMemsetAsync(h16, 0, (size_t)rows * 128 * 2, stream);
      for (int t = 0; t < TSEQ; ++t) {
        gemm(xin + (long)t * xs, xlda, wih, 384, g.bih[layer], gi, 384, rows, xk, 0);
        gemm(h16, 128, whh, 384, g.bhh[layer], gh, 384, rows, 128, 0);
        k_gru_gates<<<gblk((long)rows * 128), 256, 0, stream>>>(gi, gh, hbuf, h16, hseq, rows, t, TSEQ);
      }
      xin = hseq; xk = 128; xs = 128; xlda = (long)TSEQ * 128;
    }
    k_relu_f16<<<gblk((long)rows * 128), 256, 0, stream>>>(hbuf, h16, (long)rows * 128);
    gemm(h16, 128, W16(g.fc.w, 128, 32), 32, g.fc.b, o32, 32, rows, 128, 0);
  };

  // ================= Phase 1: encoder PropNet over 400 fused graphs =================
  f16* kp16 = (f16*)alloc(RN * 2 * 2);
  k_f32_to_f16<<<gblk(RN * 2), 256, 0, stream>>>(kp, kp16, RN * 2);
  gemm(kp16, 2, W16(enc.ne.w, 2, 96), 96, enc.ne.b, BB, 96, RN, 2, 0);
  colbn(BB, RN, enc.ne_bn, 1, ne16, nullptr);

  k_build_edge4<<<gblk(400L * 256), 256, 0, stream>>>(kp, BA);
  gemm(BA, 4, W16(enc.ee_l[0].w, 4, 96), 96, enc.ee_l[0].b, BB, 96, RE, 4, 0);
  colbn(BB, RE, enc.ee_bn[0], 1, eenc16, nullptr);

  // pstep = 1
  k_concat3<<<gblk(RE * 288), 256, 0, stream>>>(ne16, eenc16, BA, 400, 96);
  mlp2(BA, RE, 288, enc.ep_l1[0][0], enc.ep_l2[0][0], enc.ep_bn[0][0], eeff16);
  k_concat_np<<<gblk(RN * 288), 256, 0, stream>>>(ne16, ne16, eeff16, BA, 400);
  mlp2(BA, RN, 288, enc.np_l1, enc.np_l2, enc.np_bn, neff16);

  k_concat2<<<gblk(RN * 192), 256, 0, stream>>>(neff16, ne16, BA, RN, 96);
  predf(BA, RN, enc.npred_l1, enc.npred_bn, enc.npred_l2, 32, nrep, 32);
  k_concat2<<<gblk(RE * 192), 256, 0, stream>>>(eeff16, eenc16, BA, RE, 96);
  predf(BA, RE, enc.epred_l1, enc.epred_bn, enc.epred_l2, 32, erep, 32);

  // ================= Phase 2: build GRU sequence inputs =================
  k_build_node_in<<<gblk(128L * TSEQ * 40), 256, 0, stream>>>(nrep, kp, action, nin16);
  k_build_edge_in<<<gblk(2048L * TSEQ * 40), 256, 0, stream>>>(erep, kp, action, ein16);

  // ================= Phase 3: GRUs =================
  rungru(nin16, 128, 36, 40, ngru, nhseq, nagg);
  rungru(ein16, 2048, 40, 40, egru, ehseq, eagg);
  k_f32_to_f16<<<gblk(128L * 32), 256, 0, stream>>>(nagg, nagg16, 128L * 32);
  k_f32_to_f16<<<gblk(2048L * 32), 256, 0, stream>>>(eagg, eagg16, 2048L * 32);

  // ================= Phase 4: affinity PropNet (pstep=2) + gumbel softmax =================
  gemm(nagg16, 32, W16(affi.ne.w, 32, 96), 96, affi.ne.b, BB, 96, 128, 32, 0);
  colbn(BB, 128, affi.ne_bn, 1, a_ne16, nullptr);

  k_concat3<<<gblk(2048L * 96), 256, 0, stream>>>(nagg16, eagg16, eein16, 8, 32);
  gemm(eein16, 96, W16(affi.ee_l[0].w, 96, 96), 96, affi.ee_l[0].b, BB, 96, 2048, 96, 0);
  colbn(BB, 2048, affi.ee_bn[0], 1, a_eenc, nullptr);
  hipMemcpyAsync(a_eeff, a_eenc, 2048L * 96 * 2, hipMemcpyDeviceToDevice, stream);

  f16* cur_neff = a_ne16;
  for (int i = 0; i < 2; ++i) {
    k_concat3<<<gblk(2048L * 288), 256, 0, stream>>>(cur_neff, a_eeff, BA, 8, 96);
    mlp2(BA, 2048, 288, affi.ep_l1[i][0], affi.ep_l2[i][0], affi.ep_bn[i][0], a_eeff);
    k_concat_np<<<gblk(128L * 288), 256, 0, stream>>>(a_ne16, cur_neff, a_eeff, BA, 8);
    mlp2(BA, 128, 288, affi.np_l1, affi.np_l2, affi.np_bn, a_neff);
    cur_neff = a_neff;
  }
  k_concat2<<<gblk(2048L * 192), 256, 0, stream>>>(a_eeff, a_eenc, BA, 2048, 96);
  predf(BA, 2048, affi.epred_l1, affi.epred_bn, affi.epred_l2, 3, lraw, 3);
  k_gumbel<<<gblk(2048), 256, 0, stream>>>(lraw, out_lg, out_et);

  // ================= Phase 5: attribute PropNet (etn=3, edge-type weighted) =================
  gemm(nagg16, 32, W16(attr.ne.w, 32, 96), 96, attr.ne.b, BB, 96, 128, 32, 0);
  colbn(BB, 128, attr.ne_bn, 1, t_ne16, nullptr);

  f16* eks[3] = {ek0, ek1, ek2};
  for (int k = 0; k < 3; ++k) {
    gemm(eein16, 96, W16(attr.ee_l[k].w, 96, 96), 96, attr.ee_l[k].b, BB, 96, 2048, 96, 0);
    colbn(BB, 2048, attr.ee_bn[k], 1, eks[k], nullptr);
  }
  k_wsum3<<<gblk(2048L * 96), 256, 0, stream>>>(ek0, ek1, ek2, out_et, t_eenc);

  // pstep = 1
  k_concat3<<<gblk(2048L * 288), 256, 0, stream>>>(t_ne16, t_eenc, BA, 8, 96);
  for (int k = 0; k < 3; ++k)
    mlp2(BA, 2048, 288, attr.ep_l1[0][k], attr.ep_l2[0][k], attr.ep_bn[0][k], eks[k]);
  k_wsum3<<<gblk(2048L * 96), 256, 0, stream>>>(ek0, ek1, ek2, out_et, t_eeff);
  k_concat_np<<<gblk(128L * 288), 256, 0, stream>>>(t_ne16, t_ne16, t_eeff, BA, 8);
  mlp2(BA, 128, 288, attr.np_l1, attr.np_l2, attr.np_bn, t_neff);

  k_concat2<<<gblk(128L * 192), 256, 0, stream>>>(t_neff, t_ne16, BA, 128, 96);
  predf(BA, 128, attr.npred_l1, attr.npred_bn, attr.npred_l2, 4, out_na, 4);
  k_concat2<<<gblk(2048L * 192), 256, 0, stream>>>(t_eeff, t_eenc, BA, 2048, 96);
  predf(BA, 2048, attr.epred_l1, attr.epred_bn, attr.epred_l2, 4, eraw, 4);
  k_sym4<<<gblk(8L * 256 * 4), 256, 0, stream>>>(eraw, out_ea);
}
